// VQVAE_37512244363341
// MI455X (gfx1250) — matrix operand order repaired and checked by reading, not compiled, not run on hardware
//
#include <hip/hip_runtime.h>
#include <hip/hip_bf16.h>

// ---------------------------------------------------------------------------
// VQ-VAE forward on gfx1250: bf16 WMMA implicit-GEMM for conv2 / VQ / convT1.
// B-weight slabs staged into double-buffered LDS by the Tensor Data Mover
// (tensor_load_to_lds + s_wait_tensorcnt), DMA overlapped with WMMA compute.
// VQ GEMM double-buffers B fragments in registers to hide VMEM latency.
// ---------------------------------------------------------------------------

typedef __attribute__((ext_vector_type(16))) __bf16 v16bf;
typedef __attribute__((ext_vector_type(8)))  float  v8f;
typedef __attribute__((ext_vector_type(4)))  unsigned int u32x4;
typedef __attribute__((ext_vector_type(8)))  int i32x8;
typedef __attribute__((ext_vector_type(4)))  int i32x4;

union Frag { uint4 q[2]; v16bf v; };
union Q8   { uint4 q;    __bf16 h[8]; };

// ---- workspace layout (bytes); total ~105.5 MB ----------------------------
#define OFF_H1   0u               // 8*128*128*256 bf16 (encoder h; aliased by decoder d)
#define OFF_ZE   67108864u        // 32768*256 bf16 (z_e, NHWC-flat)
#define OFF_ZQ   83886080u        // 32768*256 bf16 (z_q, NHWC-flat)
#define OFF_B2   100663296u       // 4096*256 bf16 swizzled (conv2 weights)
#define OFF_BT1  102760448u       // 4*1024*256 bf16 swizzled (convT1, per-parity)
#define OFF_CB   104857600u       // 8*64*32*16 bf16 swizzled codebook^T
#define OFF_B1   105381888u       // 32*256 bf16 swizzled (conv1, K padded 16->32)
#define OFF_W2   105398272u       // 16*256 bf16 (convT2 weights, tap-major)
#define OFF_CN   105406464u       // 1024 f32 codebook norms
#define OFF_IDX  105410560u       // 32768 i32 nearest-code indices

#define PREP_TOTAL 2372608u

// ---------------------------------------------------------------------------
// Weight prep: convert to bf16 and pre-swizzle into WMMA B-fragment order:
//   Bswz[kt][nt][lane][e] = B[kt*32 + (e<8?0:16) + (lane/16)*8 + e%8][nt*16 + lane%16]
// ---------------------------------------------------------------------------
__global__ __launch_bounds__(256) void prep_kernel(
    const float* __restrict__ e1_w, const float* __restrict__ e2_w,
    const float* __restrict__ codebook, const float* __restrict__ d1_w,
    const float* __restrict__ d2_w, unsigned char* __restrict__ ws)
{
  unsigned t = blockIdx.x * 256u + threadIdx.x;
  if (t >= PREP_TOTAL) return;
  if (t < 1048576u) {
    __bf16* B2 = (__bf16*)(ws + OFF_B2);
    unsigned e = t & 15u, lane = (t >> 4) & 31u, nt = (t >> 9) & 15u, kt = t >> 13;
    unsigned k = kt * 32u + ((e & 8u) << 1) + (lane >> 4) * 8u + (e & 7u);
    unsigned n = nt * 16u + (lane & 15u);
    unsigned tap = k >> 8, ic = k & 255u, kh = tap >> 2, kw = tap & 3u;
    B2[t] = (__bf16)e2_w[((n * 256u + ic) * 4u + kh) * 4u + kw];
  } else if (t < 2097152u) {
    __bf16* BT1 = (__bf16*)(ws + OFF_BT1);
    unsigned u = t - 1048576u;
    unsigned e = u & 15u, lane = (u >> 4) & 31u, nt = (u >> 9) & 15u;
    unsigned kt = (u >> 13) & 31u, p = u >> 18;
    unsigned k = kt * 32u + ((e & 8u) << 1) + (lane >> 4) * 8u + (e & 7u);
    unsigned n = nt * 16u + (lane & 15u);
    unsigned tap = k >> 8, ic = k & 255u, th = tap >> 1, tw = tap & 1u;
    unsigned ph = p >> 1, pw = p & 1u;
    unsigned kh = ph ? (th ? 2u : 0u) : (th ? 3u : 1u);
    unsigned kw = pw ? (tw ? 2u : 0u) : (tw ? 3u : 1u);
    BT1[u] = (__bf16)d1_w[((ic * 256u + n) * 4u + kh) * 4u + kw];
  } else if (t < 2359296u) {
    __bf16* CB = (__bf16*)(ws + OFF_CB);
    unsigned u = t - 2097152u;
    unsigned e = u & 15u, lane = (u >> 4) & 31u, nt = (u >> 9) & 63u, kt = u >> 15;
    unsigned k = kt * 32u + ((e & 8u) << 1) + (lane >> 4) * 8u + (e & 7u);
    unsigned n = nt * 16u + (lane & 15u);
    CB[u] = (__bf16)codebook[n * 256u + k];
  } else if (t < 2367488u) {
    __bf16* B1 = (__bf16*)(ws + OFF_B1);
    unsigned u = t - 2359296u;
    unsigned e = u & 15u, lane = (u >> 4) & 31u, nt = u >> 9;
    unsigned k = ((e & 8u) << 1) + (lane >> 4) * 8u + (e & 7u);
    unsigned n = nt * 16u + (lane & 15u);
    B1[u] = (k < 16u) ? (__bf16)e1_w[n * 16u + k] : (__bf16)0.0f;
  } else if (t < 2371584u) {
    __bf16* W2 = (__bf16*)(ws + OFF_W2);
    unsigned u = t - 2367488u;
    unsigned tap = u >> 8, ic = u & 255u;
    W2[u] = (__bf16)d2_w[ic * 16u + tap];
  } else {
    float* CN = (float*)(ws + OFF_CN);
    unsigned n = t - 2371584u;
    float s = 0.0f;
    for (int i = 0; i < 256; ++i) { float v = codebook[n * 256u + i]; s += v * v; }
    CN[n] = s;
  }
}

// ---------------------------------------------------------------------------
// im2col A-fragment loaders (ISA 16x32 bf16 A layout:
//   element e of lane L holds K = (e<8?0:16) + (L/16)*8 + e%8, row M = L%16)
// ---------------------------------------------------------------------------
template<int LAYER>
__device__ __forceinline__ v16bf load_a(const float* __restrict__ x,
                                        const __bf16* __restrict__ act,
                                        unsigned m, unsigned kt, unsigned lane)
{
  Frag f;
  if constexpr (LAYER == 0) {
    f.q[0] = make_uint4(0u, 0u, 0u, 0u);
    f.q[1] = make_uint4(0u, 0u, 0u, 0u);
    unsigned b = m >> 14, oh = (m >> 7) & 127u, ow = m & 127u;
    unsigned kbase = (lane >> 4) * 8u;
#pragma unroll
    for (int e = 0; e < 8; ++e) {
      unsigned k = kbase + (unsigned)e;              // 0..15
      int kh = (int)(k >> 2), kw = (int)(k & 3u);
      int ih = 2 * (int)oh - 1 + kh, iw = 2 * (int)ow - 1 + kw;
      float v = (ih >= 0 && ih < 256 && iw >= 0 && iw < 256)
                    ? x[((unsigned)b * 256u + (unsigned)ih) * 256u + (unsigned)iw] : 0.0f;
      f.v[e] = (__bf16)v;
    }
  } else if constexpr (LAYER == 1) {
    unsigned b = m >> 12, oh = (m >> 6) & 63u, ow = m & 63u;
#pragma unroll
    for (int half = 0; half < 2; ++half) {
      unsigned k0 = kt * 32u + (lane >> 4) * 8u + (unsigned)half * 16u;
      unsigned tap = k0 >> 8, ic = k0 & 255u;
      int kh = (int)(tap >> 2), kw = (int)(tap & 3u);
      int ih = 2 * (int)oh - 1 + kh, iw = 2 * (int)ow - 1 + kw;
      f.q[half] = (ih >= 0 && ih < 128 && iw >= 0 && iw < 128)
          ? *(const uint4*)(act + ((((unsigned)b * 128u + (unsigned)ih) * 128u + (unsigned)iw) << 8) + ic)
          : make_uint4(0u, 0u, 0u, 0u);
    }
  } else {
    unsigned b = m >> 14, ph = (m >> 13) & 1u, pw = (m >> 12) & 1u;
    unsigned oh2 = (m >> 6) & 63u, ow2 = m & 63u;
    int oh = 2 * (int)oh2 + (int)ph, ow = 2 * (int)ow2 + (int)pw;
#pragma unroll
    for (int half = 0; half < 2; ++half) {
      unsigned k0 = kt * 32u + (lane >> 4) * 8u + (unsigned)half * 16u;  // < 1024
      unsigned tap = k0 >> 8, ic = k0 & 255u;
      int th = (int)(tap >> 1), tw = (int)(tap & 1u);
      int kh = ph ? (th ? 2 : 0) : (th ? 3 : 1);
      int kw = pw ? (tw ? 2 : 0) : (tw ? 3 : 1);
      int ih = (oh + 1 - kh) >> 1;   // exact even division (parity matched)
      int iw = (ow + 1 - kw) >> 1;
      f.q[half] = (ih >= 0 && ih < 64 && iw >= 0 && iw < 64)
          ? *(const uint4*)(act + ((((unsigned)b * 64u + (unsigned)ih) * 64u + (unsigned)iw) << 8) + ic)
          : make_uint4(0u, 0u, 0u, 0u);
    }
  }
  return f.v;
}

// ---------------------------------------------------------------------------
// TDM: issue one 16 KB B-slab copy (contiguous in global) into LDS.
// 1-D descriptor, data_size=8B, tile_dim0 = 2048 elements.
// clang-23 6-arg form: (g0, g1, g2, g3, g4, cpol). No wait here.
// ---------------------------------------------------------------------------
__device__ __forceinline__ void tdm_issue_slab(const void* gptr, unsigned ldsoff)
{
  unsigned long long ga = (unsigned long long)gptr;
  u32x4 g0;
  g0[0] = 1u;                                         // count=1, user mode
  g0[1] = ldsoff;                                     // lds_addr (bytes)
  g0[2] = (unsigned)(ga & 0xffffffffu);               // global_addr[31:0]
  g0[3] = (unsigned)((ga >> 32) & 0x1ffffffu) | (2u << 30);  // addr[56:32] | type=2
  i32x8 g1;
  g1[0] = (int)(3u << 16);                            // wg_mask=0, data_size=3 (8B)
  g1[1] = (int)(2048u << 16);                         // tensor_dim0[15:0]=2048
  g1[2] = (int)(1u << 16);                            // dim0 hi=0, tensor_dim1 lo=1
  g1[3] = (int)(2048u << 16);                         // dim1 hi=0, tile_dim0=2048
  g1[4] = 0;                                          // tile_dim1=0, tile_dim2=0
  g1[5] = 2048;                                       // tensor_dim0_stride lo
  g1[6] = 0;
  g1[7] = 0;
  i32x4 gz4 = {0, 0, 0, 0};
  i32x8 gz8 = {0, 0, 0, 0, 0, 0, 0, 0};
  __builtin_amdgcn_tensor_load_to_lds(g0, g1, gz4, gz4, gz8, 0);
}

// ---------------------------------------------------------------------------
// Implicit-GEMM conv kernel. 8 waves/block; wave owns 16(M) x 256(N) output.
// B slabs TDM-staged into double-buffered LDS; DMA for ktile kt+1 overlaps
// the 16 WMMAs + A-loads of ktile kt.
// ---------------------------------------------------------------------------
template<int LAYER, int KT>
__global__ __launch_bounds__(256) void gemm_kernel(
    const float* __restrict__ xin, const __bf16* __restrict__ act_in,
    const __bf16* __restrict__ Bsw_all, const float* __restrict__ bias,
    __bf16* __restrict__ out)
{
  __shared__ uint4 lds[2048];                       // 2 x 16 KB B slabs
  unsigned tid = threadIdx.x, lane = tid & 31u, wave = tid >> 5;
  unsigned m_block = blockIdx.x * 128u;
  const __bf16* Bsw = Bsw_all;
  if constexpr (LAYER == 2) Bsw += ((m_block >> 12) & 3u) * 262144u;   // parity slab
  unsigned m0 = m_block + wave * 16u + (lane & 15u);
  unsigned ldsbase = (unsigned)(unsigned long long)(void*)&lds[0];

  __builtin_prefetch(Bsw, 0, 0);                    // global_prefetch_b8

  v8f z8 = {0.f, 0.f, 0.f, 0.f, 0.f, 0.f, 0.f, 0.f};
  v8f acc[16];
#pragma unroll
  for (int i = 0; i < 16; ++i) acc[i] = z8;

  if (wave == 0) tdm_issue_slab((const void*)Bsw, ldsbase);   // prologue: kt=0

  for (unsigned kt = 0; kt < (unsigned)KT; ++kt) {
    unsigned buf = kt & 1u;
    if (wave == 0) __builtin_amdgcn_s_wait_tensorcnt(0);  // slab kt landed
    __syncthreads();                                 // visible to all waves;
                                                     // prior reads of other buf done
    if (wave == 0 && kt + 1u < (unsigned)KT)
      tdm_issue_slab((const void*)((const char*)Bsw + (unsigned long long)(kt + 1u) * 16384ull),
                     ldsbase + ((buf ^ 1u) << 14));   // overlap DMA with compute
    const uint4* slab = lds + buf * 1024u;
    v16bf a = load_a<LAYER>(xin, act_in, m0, kt, lane);
#pragma unroll
    for (int nt = 0; nt < 16; nt += 2) {
      Frag bfa, bfb;                                 // paired ds_load_b128s
      bfa.q[0] = slab[((unsigned)nt * 32u + lane) * 2u];
      bfa.q[1] = slab[((unsigned)nt * 32u + lane) * 2u + 1u];
      bfb.q[0] = slab[((unsigned)(nt + 1) * 32u + lane) * 2u];
      bfb.q[1] = slab[((unsigned)(nt + 1) * 32u + lane) * 2u + 1u];
      acc[nt] = __builtin_amdgcn_wmma_f32_16x16x32_bf16(
          false, a, false, bfa.v, (short)0, acc[nt], false, false);
      acc[nt + 1] = __builtin_amdgcn_wmma_f32_16x16x32_bf16(
          false, a, false, bfb.v, (short)0, acc[nt + 1], false, false);
    }
  }

  // epilogue: bias + ReLU, bf16 NHWC store
  unsigned half = lane >> 4, nlo = lane & 15u;
#pragma unroll
  for (int nt = 0; nt < 16; ++nt) {
    unsigned n = (unsigned)nt * 16u + nlo;
    float bv = bias[n];
#pragma unroll
    for (int r = 0; r < 8; ++r) {
      unsigned m = m_block + wave * 16u + (unsigned)r + half * 8u;
      float v = acc[nt][r] + bv;
      v = v > 0.0f ? v : 0.0f;
      unsigned long addr;
      if constexpr (LAYER == 2) {
        unsigned b = m >> 14, ph = (m >> 13) & 1u, pw = (m >> 12) & 1u;
        unsigned oh = 2u * ((m >> 6) & 63u) + ph, ow = 2u * (m & 63u) + pw;
        addr = (unsigned long)((b * 128u + oh) * 128u + ow) * 256u + n;
      } else {
        addr = (unsigned long)m * 256u + n;        // NHWC-flat, m = (b,h,w)
      }
      out[addr] = (__bf16)v;
    }
  }
}

// ---------------------------------------------------------------------------
// VQ: dot = z_e @ codebook^T via WMMA; dist = |c|^2 - 2*dot. Double-buffered
// B fragments: loads for tile nt+1 are issued before the WMMAs of tile nt.
// ---------------------------------------------------------------------------
__global__ __launch_bounds__(256) void vq_kernel(
    const __bf16* __restrict__ ze, const __bf16* __restrict__ cbs,
    const float* __restrict__ cnorm, int* __restrict__ idx_out)
{
  unsigned tid = threadIdx.x, lane = tid & 31u, wave = tid >> 5;
  unsigned m0 = blockIdx.x * 128u + wave * 16u;
  unsigned row = lane & 15u, half = lane >> 4;

  Frag a[8];                                        // full K=256 A fragments
  const uint4* zeq = (const uint4*)(ze + (unsigned long)(m0 + row) * 256u);
#pragma unroll
  for (int kt = 0; kt < 8; ++kt) {
    a[kt].q[0] = zeq[(unsigned)kt * 4u + half];
    a[kt].q[1] = zeq[(unsigned)kt * 4u + half + 2u];
  }

  v8f z8 = {0.f, 0.f, 0.f, 0.f, 0.f, 0.f, 0.f, 0.f};
  float best[8]; int bidx[8];
#pragma unroll
  for (int r = 0; r < 8; ++r) { best[r] = 3.4e38f; bidx[r] = 0; }

  const uint4* Bg = (const uint4*)cbs;

  auto loadB = [&](Frag (&dst)[8], unsigned nt) {
#pragma unroll
    for (int kt = 0; kt < 8; ++kt) {
      dst[kt].q[0] = Bg[(((unsigned)kt * 64u + nt) * 32u + lane) * 2u];
      dst[kt].q[1] = Bg[(((unsigned)kt * 64u + nt) * 32u + lane) * 2u + 1u];
    }
  };
  auto process = [&](Frag (&bf)[8], unsigned nt) {
    v8f acc = z8;
#pragma unroll
    for (int kt = 0; kt < 8; ++kt)
      acc = __builtin_amdgcn_wmma_f32_16x16x32_bf16(
          false, a[kt].v, false, bf[kt].v, (short)0, acc, false, false);
    unsigned n = nt * 16u + (lane & 15u);
    float cn = cnorm[n];
#pragma unroll
    for (int r = 0; r < 8; ++r) {
      float dist = cn - 2.0f * acc[r];
      if (dist < best[r]) { best[r] = dist; bidx[r] = (int)n; }
    }
  };

  Frag b0[8], b1[8];
  loadB(b0, 0);
  for (unsigned nt = 0; nt < 64u; nt += 2u) {
    loadB(b1, nt + 1u);          // issue next tile's loads before computing
    process(b0, nt);
    if (nt + 2u < 64u) loadB(b0, nt + 2u);
    process(b1, nt + 1u);
  }

#pragma unroll
  for (int off = 1; off < 16; off <<= 1) {
#pragma unroll
    for (int r = 0; r < 8; ++r) {
      float ov = __shfl_xor(best[r], off, 32);
      int   oi = __shfl_xor(bidx[r], off, 32);
      if (ov < best[r] || (ov == best[r] && oi < bidx[r])) { best[r] = ov; bidx[r] = oi; }
    }
  }
  if ((lane & 15u) == 0u) {
#pragma unroll
    for (int r = 0; r < 8; ++r)
      idx_out[m0 + (unsigned)r + half * 8u] = bidx[r];
  }
}

// z_q gather: fp32 NCHW to d_out (coalesced: m fastest) + bf16 NHWC for decoder
__global__ __launch_bounds__(256) void gather_kernel(
    const int* __restrict__ idx, const float* __restrict__ codebook,
    float* __restrict__ zq_f32, __bf16* __restrict__ zq_bf)
{
  unsigned t = blockIdx.x * 256u + threadIdx.x;     // 8,388,608 threads
  unsigned c = t >> 15, m = t & 32767u;
  unsigned b = m >> 12, hw = m & 4095u;
  float v = codebook[(unsigned long)idx[m] * 256u + c];
  zq_f32[(unsigned long)(b * 256u + c) * 4096u + hw] = v;
  zq_bf[(unsigned long)m * 256u + c] = (__bf16)v;
}

// convT2 (256 -> 1 channel): ~1 GFLOP, VALU; d fits in L2
__global__ __launch_bounds__(256) void convt2_kernel(
    const __bf16* __restrict__ d, const __bf16* __restrict__ w2,
    const float* __restrict__ d2_b, float* __restrict__ out)
{
  unsigned t = blockIdx.x * 256u + threadIdx.x;     // 524,288 threads
  unsigned b = t >> 16, oh = (t >> 8) & 255u, ow = t & 255u;
  unsigned ph = oh & 1u, pw = ow & 1u;
  float acc = d2_b[0];
#pragma unroll
  for (int th = 0; th < 2; ++th) {
#pragma unroll
    for (int tw = 0; tw < 2; ++tw) {
      int kh = ph ? (th ? 2 : 0) : (th ? 3 : 1);
      int kw = pw ? (tw ? 2 : 0) : (tw ? 3 : 1);
      int ih = ((int)oh + 1 - kh) >> 1, iw = ((int)ow + 1 - kw) >> 1;
      if (ih >= 0 && ih < 128 && iw >= 0 && iw < 128) {
        const __bf16* dp = d + ((((unsigned)b * 128u + (unsigned)ih) * 128u + (unsigned)iw) << 8);
        const __bf16* wp = w2 + (unsigned)(kh * 4 + kw) * 256u;
        for (int ic = 0; ic < 256; ic += 8) {
          Q8 dv, wv;
          dv.q = *(const uint4*)(dp + ic);
          wv.q = *(const uint4*)(wp + ic);
#pragma unroll
          for (int j = 0; j < 8; ++j) acc += (float)dv.h[j] * (float)wv.h[j];
        }
      }
    }
  }
  out[t] = acc;
}

// ---------------------------------------------------------------------------
extern "C" void kernel_launch(void* const* d_in, const int* in_sizes, int n_in,
                              void* d_out, int out_size, void* d_ws, size_t ws_size,
                              hipStream_t stream) {
  const float* x        = (const float*)d_in[0];
  const float* e1_w     = (const float*)d_in[1];
  const float* e1_b     = (const float*)d_in[2];
  const float* e2_w     = (const float*)d_in[3];
  const float* e2_b     = (const float*)d_in[4];
  const float* codebook = (const float*)d_in[5];
  const float* d1_w     = (const float*)d_in[6];
  const float* d1_b     = (const float*)d_in[7];
  const float* d2_w     = (const float*)d_in[8];
  const float* d2_b     = (const float*)d_in[9];

  unsigned char* ws = (unsigned char*)d_ws;        // needs ~105.6 MB
  __bf16* h1   = (__bf16*)(ws + OFF_H1);           // also decoder 'd' (aliased)
  __bf16* ze   = (__bf16*)(ws + OFF_ZE);
  __bf16* zq   = (__bf16*)(ws + OFF_ZQ);
  __bf16* B2   = (__bf16*)(ws + OFF_B2);
  __bf16* BT1  = (__bf16*)(ws + OFF_BT1);
  __bf16* CB   = (__bf16*)(ws + OFF_CB);
  __bf16* B1   = (__bf16*)(ws + OFF_B1);
  __bf16* W2   = (__bf16*)(ws + OFF_W2);
  float*  CN   = (float*) (ws + OFF_CN);
  int*    IDX  = (int*)   (ws + OFF_IDX);
  float*  out  = (float*)d_out;                    // [recon 524288][z_q 8388608]

  prep_kernel<<<PREP_TOTAL / 256u, 256, 0, stream>>>(e1_w, e2_w, codebook, d1_w, d2_w, ws);
  gemm_kernel<0, 1>  <<<1024, 256, 0, stream>>>(x,       nullptr, B1,  e1_b, h1);  // conv1
  gemm_kernel<1, 128><<<256,  256, 0, stream>>>(nullptr, h1,      B2,  e2_b, ze);  // conv2
  vq_kernel          <<<256,  256, 0, stream>>>(ze, CB, CN, IDX);
  gather_kernel      <<<32768, 256, 0, stream>>>(IDX, codebook, out + 524288, zq);
  gemm_kernel<2, 32> <<<1024, 256, 0, stream>>>(nullptr, zq,      BT1, d1_b, h1);  // convT1 -> d
  convt2_kernel      <<<2048, 256, 0, stream>>>(h1, W2, d2_b, out);                // convT2
}
